// BezierReinforceWrapper_29858612642042
// MI455X (gfx1250) — matrix operand order repaired and checked by reading, not compiled, not run on hardware
//
#include <hip/hip_runtime.h>
#include <hip/hip_bf16.h>

// ---------------------------------------------------------------------------
// Problem constants (from the reference)
// ---------------------------------------------------------------------------
#define B_SZ    2048
#define D_IN    256
#define S6      192          // S * 6
#define CANVAS  128
#define NUM_T   50
#define N_PTS   (32 * NUM_T) // 1600 (s,t) points per image
#define W_STAMP (-0.07f)
#define BG      0.3f

// dim * (-log(scale) - 0.5*log(2pi)) with scale = 1e-4, dim = 192
#define LOGPROB_CONST  1591.9491530441459f
// dim * (0.5 + 0.5*log(2pi) + log(scale))
#define ENTROPY_CONST  (-1495.9491530441459f)

typedef __attribute__((ext_vector_type(16))) _Float16 v16h;
typedef __attribute__((ext_vector_type(8)))  float    v8f;

// ---------------------------------------------------------------------------
// Kernel 1: mu = x @ W + b ; sample = sigmoid(mu)
// One wave32 per 16x16 output tile. K=256 swept as 8 x wmma_f32_16x16x32_f16.
// f16 inputs / f32 accumulate: inputs are O(1) magnitudes, well within f16.
// ---------------------------------------------------------------------------
__global__ __launch_bounds__(32)
void gemm_sigmoid_kernel(const float* __restrict__ x,     // [2048, 256]
                         const float* __restrict__ W,     // [256, 192]
                         const float* __restrict__ bias,  // [192]
                         float* __restrict__ sample)      // [2048, 192]
{
    const int lane = threadIdx.x & 31;
    const int l15  = lane & 15;
    const int half = lane >> 4;          // 0: K-low half, 1: K-high half

    const int tileM = blockIdx.x;        // 0..127
    const int tileN = blockIdx.y;        // 0..11

    const int rowA = tileM * 16 + l15;   // this lane's A row
    const int colB = tileN * 16 + l15;   // this lane's B column

    v8f acc = {};

    #pragma unroll
    for (int k0 = 0; k0 < D_IN; k0 += 32) {
        v16h a, b;
        // A fragment (16x32 f16, ISA 7.12.2): lane<16 holds K {0..7,16..23},
        // lane>=16 holds K {8..15,24..31}, two f16 per VGPR.
        #pragma unroll
        for (int j = 0; j < 16; ++j) {
            const int v  = j >> 1;
            const int ka = k0 + ((v >> 2) << 4) + (half << 3)
                              + ((v & 3) << 1) + (j & 1);
            a[j] = (_Float16)x[rowA * D_IN + ka];
            // B fragment (32x16 f16): same K order as A: lanes 0-15 hold
            // K {0..7,16..23}, lanes 16-31 hold K {8..15,24..31} (N=lane&15).
            const int kb = k0 + ((j >> 3) << 4) + (half << 3) + (j & 7);
            b[j] = (_Float16)W[kb * S6 + colB];
        }
        acc = __builtin_amdgcn_wmma_f32_16x16x32_f16(
                  /*neg_a=*/false, a, /*neg_b=*/false, b,
                  /*c_mod=*/(short)0, acc,
                  /*reuse_a=*/false, /*reuse_b=*/false);
    }

    // C/D layout: VGPR r holds M = r + 8*half, N = lane&15.
    #pragma unroll
    for (int r = 0; r < 8; ++r) {
        const int m = tileM * 16 + r + (half << 3);
        const int n = tileN * 16 + l15;
        const float mu = acc[r] + bias[n];
        const float s  = 1.0f / (1.0f + __expf(-mu));
        sample[m * S6 + n] = s;
    }
}

// ---------------------------------------------------------------------------
// Kernel 2: spline painter. One workgroup per image; 128x128 f32 canvas lives
// entirely in LDS (64 KB of the WGP's 320 KB -> 5 blocks/WGP). 14400 stamps
// per image become ds_add_f32 LDS atomics instead of global RMW traffic;
// the only HBM traffic is the coalesced b128 write of the final canvas.
// ---------------------------------------------------------------------------
__global__ __launch_bounds__(256)
void paint_kernel(const float* __restrict__ sample,  // [2048, 192]
                  float* __restrict__ sketch,        // [2048, 128, 128]
                  float* __restrict__ logp,          // [2048]
                  float* __restrict__ ent)           // [2048]
{
    __shared__ float canvas[CANVAS * CANVAS];        // 64 KB

    const int b   = blockIdx.x;
    const int tid = threadIdx.x;

    // Zero the canvas with b128 LDS stores.
    {
        float4* c4 = (float4*)canvas;
        const float4 z = make_float4(0.f, 0.f, 0.f, 0.f);
        for (int i = tid; i < (CANVAS * CANVAS) / 4; i += 256) c4[i] = z;
    }
    __syncthreads();

    const float* sp = sample + (size_t)b * S6;

    // 1600 (spline, t) points striped across 256 threads; 9 stamps each.
    for (int p = tid; p < N_PTS; p += 256) {
        const int   s  = p / NUM_T;
        const int   ti = p - s * NUM_T;
        const float t  = (float)ti * (1.0f / (float)(NUM_T - 1));
        const float u  = 1.0f - t;
        const float w0 = u * u;
        const float w1 = 2.0f * u * t;
        const float w2 = t * t;

        const float px = (w0 * sp[s * 6 + 0] + w1 * sp[s * 6 + 2] +
                          w2 * sp[s * 6 + 4]) * (float)CANVAS;
        const float py = (w0 * sp[s * 6 + 1] + w1 * sp[s * 6 + 3] +
                          w2 * sp[s * 6 + 5]) * (float)CANVAS;

        const int cx = (int)rintf(px);   // round-half-even == jnp.round
        const int cy = (int)rintf(py);

        #pragma unroll
        for (int dx = -1; dx <= 1; ++dx) {
            #pragma unroll
            for (int dy = -1; dy <= 1; ++dy) {
                const int xi = min(max(cx + dx, 0), CANVAS - 1);
                const int yi = min(max(cy + dy, 0), CANVAS - 1);
                atomicAdd(&canvas[xi * CANVAS + yi], W_STAMP);  // ds_add_f32
            }
        }
    }
    __syncthreads();

    // clip(canvas + BG, 0, 1) -> coalesced b128 stream to HBM.
    {
        const float4* c4 = (const float4*)canvas;
        float4* o4 = (float4*)(sketch + (size_t)b * CANVAS * CANVAS);
        for (int i = tid; i < (CANVAS * CANVAS) / 4; i += 256) {
            float4 v = c4[i];
            v.x = fminf(fmaxf(v.x + BG, 0.0f), 1.0f);
            v.y = fminf(fmaxf(v.y + BG, 0.0f), 1.0f);
            v.z = fminf(fmaxf(v.z + BG, 0.0f), 1.0f);
            v.w = fminf(fmaxf(v.w + BG, 0.0f), 1.0f);
            o4[i] = v;
        }
    }

    // log_prob / entropy are closed-form constants (raw == mu).
    if (tid == 0) {
        logp[b] = LOGPROB_CONST;
        ent[b]  = ENTROPY_CONST;
    }
}

// ---------------------------------------------------------------------------
// Launch. d_out layout (f32, return order): sketch | log_prob | entropy | sample
// ---------------------------------------------------------------------------
extern "C" void kernel_launch(void* const* d_in, const int* in_sizes, int n_in,
                              void* d_out, int out_size, void* d_ws, size_t ws_size,
                              hipStream_t stream) {
    const float* x    = (const float*)d_in[0];   // [2048, 256]
    const float* W    = (const float*)d_in[1];   // [256, 192]
    const float* bias = (const float*)d_in[2];   // [192]

    float* out    = (float*)d_out;
    float* sketch = out;                                        // 2048*16384
    float* logp   = out + (size_t)B_SZ * CANVAS * CANVAS;       // +2048
    float* ent    = logp + B_SZ;                                // +2048
    float* sample = ent + B_SZ;                                 // +2048*192

    // GEMM + sigmoid: 128 M-tiles x 12 N-tiles, one wave each.
    dim3 g1(B_SZ / 16, S6 / 16);
    gemm_sigmoid_kernel<<<g1, 32, 0, stream>>>(x, W, bias, sample);

    // Painter: one workgroup per image, canvas resident in LDS.
    paint_kernel<<<B_SZ, 256, 0, stream>>>(sample, sketch, logp, ent);
}
